// Attention_80058190397874
// MI455X (gfx1250) — matrix operand order repaired and checked by reading, not compiled, not run on hardware
//
#include <hip/hip_runtime.h>

typedef __attribute__((ext_vector_type(16))) _Float16 v16h;
typedef __attribute__((ext_vector_type(8)))  _Float16 v8h;
typedef __attribute__((ext_vector_type(8)))  float    v8f;
typedef __attribute__((ext_vector_type(4)))  float    v4f;

#define DEVFN static __device__ __forceinline__

// ---------------- problem constants ----------------
constexpr int N_NODES = 20000;
constexpr int N_EDGES = 320000;
constexpr int FEAT    = 160;   // 64 + 32*3
constexpr int NB      = 16;
constexpr int TOPO    = 24;

// ---------------- f16 weight block layout (in halves) ----------------
constexpr size_t WOFF_Q0 = 0;          // q_W0: 64*8*32
constexpr size_t WOFF_Q1 = 16384;      // q_W1: 32*8*16
constexpr size_t WOFF_BR = 20480;      // k-branch base; v = +BR_HALVES
constexpr size_t BOFF_LIN0W0 = 0;      // 64*8*64
constexpr size_t BOFF_LIN0W1 = 32768;  // 32*8*32
constexpr size_t BOFF_LINW0  = 40960;  // 96*8*32
constexpr size_t BOFF_LINW1  = 65536;  // 96*8*16
constexpr size_t BOFF_SCW0   = 77824;  // 64*8*32
constexpr size_t BOFF_SCW1   = 94208;  // 32*8*16
constexpr size_t BOFF_FCW1   = 98304;  // 64*64 (rows>=40 zero-padded)
constexpr size_t BOFF_FCW2   = 102400; // 64*64
constexpr size_t BOFF_FCW3   = 106496; // 64*192
constexpr size_t BR_HALVES   = 118784;
constexpr size_t W_TOTAL_HALVES = WOFF_BR + 2 * BR_HALVES;   // 258048

// ---------------- workspace layout (bytes) ----------------
constexpr size_t OFF_W    = 0;
constexpr size_t OFF_QH   = W_TOTAL_HALVES * 2;                    // 516096
constexpr size_t OFF_LOG  = OFF_QH   + (size_t)N_NODES * 80 * 4;   // qh f32
constexpr size_t OFF_VH   = OFF_LOG  + (size_t)N_EDGES * 2 * 4;    // logits f32
constexpr size_t OFF_MENC = OFF_VH   + (size_t)N_EDGES * 80 * 2;   // vh f16
constexpr size_t OFF_DEN  = OFF_MENC + (size_t)N_NODES * 2 * 4;
constexpr size_t OFF_NUM  = OFF_DEN  + (size_t)N_NODES * 2 * 4;

// ---------------- small helpers ----------------
DEVFN int laneid() { return (int)(threadIdx.x & 31); }

DEVFN v8f zero8() { v8f z = {0.f,0.f,0.f,0.f,0.f,0.f,0.f,0.f}; return z; }

DEVFN v16h pack16(v8h lo, v8h hi) {
  v16h r;
#pragma unroll
  for (int i = 0; i < 8; i++) { r[i] = lo[i]; r[i + 8] = hi[i]; }
  return r;
}

DEVFN v8h ldv8(const _Float16* p) { return *(const v8h*)p; }

// A-fragment (16x32 f16, M = lane&15) from row-major f16 LDS [16][stride]
DEVFN v16h frag_a(const _Float16* base, int stride, int kbase) {
  int l = laneid(); int m = l & 15; int kh = (l >> 4) & 1;
  const _Float16* p = base + m * stride + kbase + kh * 8;
  return pack16(ldv8(p), ldv8(p + 16));     // e<8: K=kh*8+e (+kt*32); e>=8: +16
}

// B-fragment (32x16 f16, N = lane&15): e<8: K=kh*8+e; e>=8: K=16+kh*8+(e-8); row K at p + K*ks
DEVFN v16h frag_b(const _Float16* p, int ks) {
  int l = laneid(); int n = l & 15; int kh = (l >> 4) & 1; v16h r;
#pragma unroll
  for (int e = 0; e < 16; e++) { int k = (e < 8) ? (kh * 8 + e) : (16 + kh * 8 + e - 8); r[e] = p[(size_t)k * ks + n]; }
  return r;
}

DEVFN v8f wmma16(v16h a, v16h b, v8f c) {
  return __builtin_amdgcn_wmma_f32_16x16x32_f16(false, a, false, b, (short)0, c, false, false);
}

// C/D fragment scatter: N = lane&15 (+nbase), rows m = (lane>>4)*8 + e
DEVFN void store_c_h(_Float16* base, int stride, int nbase, v8f c) {
  int l = laneid(); int n = nbase + (l & 15); int mo = (l >> 4) * 8;
#pragma unroll
  for (int e = 0; e < 8; e++) base[(mo + e) * stride + n] = (_Float16)c[e];
}
DEVFN void store_c_f(float* base, int stride, int nbase, v8f c) {
  int l = laneid(); int n = nbase + (l & 15); int mo = (l >> 4) * 8;
#pragma unroll
  for (int e = 0; e < 8; e++) base[(mo + e) * stride + n] = c[e];
}

DEVFN void silu8(v8f& c) {
#pragma unroll
  for (int e = 0; e < 8; e++) { float x = c[e]; c[e] = x / (1.f + __expf(-x)); }
}

// monotonic float<->uint encoding for atomicMax on floats
DEVFN unsigned fenc(float f) { unsigned u = __float_as_uint(f); return (u & 0x80000000u) ? ~u : (u | 0x80000000u); }
DEVFN float    fdec(unsigned u) { return (u & 0x80000000u) ? __uint_as_float(u ^ 0x80000000u) : __uint_as_float(~u); }

// ---------------- weight conversion (f32 -> f16, norm folded) ----------------
__global__ void convert_kernel(const float* __restrict__ src, _Float16* __restrict__ dst,
                               int rows, int cols, int srcRows, float scale) {
  int i = blockIdx.x * blockDim.x + threadIdx.x;
  if (i >= rows * cols) return;
  int r = i / cols;
  dst[i] = (r < srcRows) ? (_Float16)(src[i] * scale) : (_Float16)0.f;
}

// ---------------- q kernel: per-node fctp via WMMA ----------------
struct alignas(64) QScr {
  float    attr[16 * 8];
  _Float16 x0[16 * 64];
  _Float16 x1[3 * 16 * 32];
  float    q0[16 * 32];
  float    q1[3 * 16 * 16];
};

__global__ __launch_bounds__(128) void q_kernel(const float* __restrict__ feats,
                                                const float* __restrict__ attrs,
                                                const _Float16* __restrict__ W,
                                                float* __restrict__ qh) {
  __shared__ QScr S[4];
  const int wave = threadIdx.x >> 5, lane = threadIdx.x & 31;
  QScr& s = S[wave];
  int tile = blockIdx.x * 4 + wave;
  const int nTiles = N_NODES / 16;
  const bool active = tile < nTiles;
  if (!active) tile = nTiles - 1;
  const int n0 = tile * 16;

  for (int i = lane; i < 16 * 64; i += 32) { int m = i >> 6, u = i & 63; s.x0[i] = (_Float16)feats[(size_t)(n0 + m) * FEAT + u]; }
  for (int i = lane; i < 3 * 16 * 32; i += 32) { int c = i >> 9, r = i & 511, m = r >> 5, u = r & 31; s.x1[i] = (_Float16)feats[(size_t)(n0 + m) * FEAT + 64 + u * 3 + c]; }
  for (int i = lane; i < 16 * 8; i += 32) { int m = i >> 3, a = i & 7; s.attr[i] = attrs[(size_t)(n0 + m) * 8 + a]; }
  __syncthreads();

  const int m = lane & 15;
  float attrRow[8];
  { v4f a0 = *(const v4f*)&s.attr[m * 8]; v4f a1 = *(const v4f*)&s.attr[m * 8 + 4];
#pragma unroll
    for (int i = 0; i < 4; i++) { attrRow[i] = a0[i]; attrRow[4 + i] = a1[i]; } }

  v16h ax00 = frag_a(s.x0, 64, 0), ax01 = frag_a(s.x0, 64, 32);
  v16h ax1[3];
#pragma unroll
  for (int c = 0; c < 3; c++) ax1[c] = frag_a(s.x1 + c * 512, 32, 0);

  v8f q0acc[2] = { zero8(), zero8() };
  v8f q1acc[3] = { zero8(), zero8(), zero8() };
#pragma unroll
  for (int a = 0; a < 8; a++) {
    _Float16 sa = (_Float16)attrRow[a];
    v16h t0 = ax00 * sa, t1 = ax01 * sa;
#pragma unroll
    for (int nt = 0; nt < 2; nt++) {
      q0acc[nt] = wmma16(t0, frag_b(W + WOFF_Q0 + ((size_t)0 * 8 + a) * 32 + nt * 16, 256), q0acc[nt]);
      q0acc[nt] = wmma16(t1, frag_b(W + WOFF_Q0 + ((size_t)(32) * 8 + a) * 32 + nt * 16, 256), q0acc[nt]);
    }
#pragma unroll
    for (int c = 0; c < 3; c++) {
      v16h tc = ax1[c] * sa;
      q1acc[c] = wmma16(tc, frag_b(W + WOFF_Q1 + ((size_t)0 * 8 + a) * 16, 128), q1acc[c]);
    }
  }
#pragma unroll
  for (int nt = 0; nt < 2; nt++) store_c_f(s.q0, 32, nt * 16, q0acc[nt]);
#pragma unroll
  for (int c = 0; c < 3; c++) store_c_f(s.q1 + c * 256, 16, 0, q1acc[c]);
  __syncthreads();

  for (int i = lane; i < 16 * 80; i += 32) {
    int m2 = i / 80, f = i % 80, h = f / 40, p = f % 40;
    float val;
    if (p < 16) val = s.q0[m2 * 32 + h * 16 + p];
    else { int t = p - 16; int f1 = h * 24 + t; int w = f1 / 3, c = f1 % 3; val = s.q1[c * 256 + m2 * 16 + w]; }
    if (active) qh[(size_t)(n0 + m2) * 80 + f] = val;
  }
}

// ---------------- edge kernel: k (logits) / v (store vh) ----------------
struct alignas(64) EdgeScr {
  int      srcI[16];
  int      dstI[16];
  float    attr[16 * 8];
  float    sh[16 * 4];
  _Float16 x0[16 * 64];
  _Float16 x1[3 * 16 * 32];
  _Float16 radh[16 * 64];
  _Float16 hh[16 * 64];
  _Float16 wmlp[16 * 192];
  _Float16 l0[16 * 64];
  _Float16 l1[3 * 16 * 32];
  float    o0f[16 * 32];
  float    o1f[3 * 16 * 16];
};

template<bool IS_V>
__global__ __launch_bounds__(128) void edge_kv_kernel(
    const float* __restrict__ feats, const float* __restrict__ attrs,
    const int* __restrict__ esrc, const int* __restrict__ edst,
    const float* __restrict__ esh, const float* __restrict__ erad, const float* __restrict__ etopo,
    const _Float16* __restrict__ W, const float* __restrict__ qh,
    unsigned* __restrict__ mEnc, float* __restrict__ logits, _Float16* __restrict__ vh) {
  __shared__ EdgeScr S[4];
  const int wave = threadIdx.x >> 5, lane = threadIdx.x & 31;
  EdgeScr& s = S[wave];
  int tile = blockIdx.x * 4 + wave;
  const int nTiles = N_EDGES / 16;
  const bool active = tile < nTiles;
  if (!active) tile = nTiles - 1;
  const int e0 = tile * 16;

  if (lane < 16) { s.srcI[lane] = esrc[e0 + lane]; s.dstI[lane] = edst[e0 + lane]; }
  __syncthreads();

  for (int i = lane; i < 16 * 64; i += 32) { int m = i >> 6, u = i & 63; s.x0[i] = (_Float16)feats[(size_t)s.srcI[m] * FEAT + u]; }
  for (int i = lane; i < 3 * 16 * 32; i += 32) { int c = i >> 9, r = i & 511, m = r >> 5, u = r & 31; s.x1[i] = (_Float16)feats[(size_t)s.srcI[m] * FEAT + 64 + u * 3 + c]; }
  for (int i = lane; i < 16 * 8; i += 32) { int m = i >> 3, a = i & 7; s.attr[i] = attrs[(size_t)s.srcI[m] * 8 + a]; }
  for (int i = lane; i < 16 * 4; i += 32) { int m = i >> 2; s.sh[i] = esh[(size_t)(e0 + m) * 4 + (i & 3)]; }
  for (int i = lane; i < 16 * 64; i += 32) {
    int m = i >> 6, j = i & 63; float v = 0.f;
    if (j < NB) v = erad[(size_t)(e0 + m) * NB + j];
    else if (j < NB + TOPO) v = etopo[(size_t)(e0 + m) * TOPO + (j - NB)];
    s.radh[i] = (_Float16)v;
  }
  __syncthreads();

  const int m = lane & 15, kh = lane >> 4;
  float attrRow[8];
  { v4f a0 = *(const v4f*)&s.attr[m * 8]; v4f a1 = *(const v4f*)&s.attr[m * 8 + 4];
#pragma unroll
    for (int i = 0; i < 4; i++) { attrRow[i] = a0[i]; attrRow[4 + i] = a1[i]; } }
  const float sh0 = s.sh[m * 4];
  float sh1v[3] = { s.sh[m * 4 + 1], s.sh[m * 4 + 2], s.sh[m * 4 + 3] };

  // -------- radial MLP: 40(->64 pad) -> 64 -> 64 -> 192 --------
  {
    v16h a0 = frag_a(s.radh, 64, 0), a1 = frag_a(s.radh, 64, 32);
#pragma unroll
    for (int nt = 0; nt < 4; nt++) {
      v8f acc = zero8();
      acc = wmma16(a0, frag_b(W + BOFF_FCW1 + (size_t)0 * 64 + nt * 16, 64), acc);
      acc = wmma16(a1, frag_b(W + BOFF_FCW1 + (size_t)(32) * 64 + nt * 16, 64), acc);
      silu8(acc);
      store_c_h(s.hh, 64, nt * 16, acc);
    }
  }
  __syncthreads();
  {
    v16h a0 = frag_a(s.hh, 64, 0), a1 = frag_a(s.hh, 64, 32);
    __syncthreads();                      // all lanes read hh before overwrite
#pragma unroll
    for (int nt = 0; nt < 4; nt++) {
      v8f acc = zero8();
      acc = wmma16(a0, frag_b(W + BOFF_FCW2 + (size_t)0 * 64 + nt * 16, 64), acc);
      acc = wmma16(a1, frag_b(W + BOFF_FCW2 + (size_t)(32) * 64 + nt * 16, 64), acc);
      silu8(acc);
      store_c_h(s.hh, 64, nt * 16, acc);
    }
  }
  __syncthreads();
  {
    v16h a0 = frag_a(s.hh, 64, 0), a1 = frag_a(s.hh, 64, 32);
#pragma unroll
    for (int nt = 0; nt < 12; nt++) {
      v8f acc = zero8();
      acc = wmma16(a0, frag_b(W + BOFF_FCW3 + (size_t)0 * 192 + nt * 16, 192), acc);
      acc = wmma16(a1, frag_b(W + BOFF_FCW3 + (size_t)(32) * 192 + nt * 16, 192), acc);
      store_c_h(s.wmlp, 192, nt * 16, acc);
    }
  }

  // -------- lin0 fctp: l0 (16x64), l1 (16x32 per component) --------
  v16h ax00 = frag_a(s.x0, 64, 0), ax01 = frag_a(s.x0, 64, 32);
  v16h ax1[3];
#pragma unroll
  for (int c = 0; c < 3; c++) ax1[c] = frag_a(s.x1 + c * 512, 32, 0);
  {
    v8f l0acc[4]; v8f l1acc[3][2];
#pragma unroll
    for (int nt = 0; nt < 4; nt++) l0acc[nt] = zero8();
#pragma unroll
    for (int c = 0; c < 3; c++) { l1acc[c][0] = zero8(); l1acc[c][1] = zero8(); }
#pragma unroll
    for (int a = 0; a < 8; a++) {
      _Float16 sa = (_Float16)attrRow[a];
      v16h t0 = ax00 * sa, t1 = ax01 * sa;
#pragma unroll
      for (int nt = 0; nt < 4; nt++) {
        l0acc[nt] = wmma16(t0, frag_b(W + BOFF_LIN0W0 + ((size_t)0 * 8 + a) * 64 + nt * 16, 512), l0acc[nt]);
        l0acc[nt] = wmma16(t1, frag_b(W + BOFF_LIN0W0 + ((size_t)(32) * 8 + a) * 64 + nt * 16, 512), l0acc[nt]);
      }
#pragma unroll
      for (int c = 0; c < 3; c++) {
        v16h tc = ax1[c] * sa;
#pragma unroll
        for (int nt = 0; nt < 2; nt++)
          l1acc[c][nt] = wmma16(tc, frag_b(W + BOFF_LIN0W1 + ((size_t)0 * 8 + a) * 32 + nt * 16, 256), l1acc[c][nt]);
      }
    }
#pragma unroll
    for (int nt = 0; nt < 4; nt++) store_c_h(s.l0, 64, nt * 16, l0acc[nt]);
#pragma unroll
    for (int c = 0; c < 3; c++) { store_c_h(s.l1 + c * 512, 32, 0, l1acc[c][0]); store_c_h(s.l1 + c * 512, 32, 16, l1acc[c][1]); }
  }
  __syncthreads();

  // -------- build mid fragments on the fly (no staging) --------
  auto mid0seg = [&](int j0) -> v8h {
    if (j0 < 64) {
      return ldv8(&s.l0[m * 64 + j0]) * ldv8(&s.wmlp[m * 192 + j0]) * (_Float16)sh0;
    } else {
      int u = j0 - 64;
      v8h sum = ldv8(&s.l1[0 * 512 + m * 32 + u]) * (_Float16)sh1v[0];
      sum = sum + ldv8(&s.l1[1 * 512 + m * 32 + u]) * (_Float16)sh1v[1];
      sum = sum + ldv8(&s.l1[2 * 512 + m * 32 + u]) * (_Float16)sh1v[2];
      return sum * ldv8(&s.wmlp[m * 192 + 160 + u]) * (_Float16)0.57735027f; // 1/sqrt(3)
    }
  };
  auto mid1seg = [&](int c, int j0) -> v8h {
    if (j0 < 64) {
      return ldv8(&s.l0[m * 64 + j0]) * (_Float16)sh1v[c] * ldv8(&s.wmlp[m * 192 + 64 + j0]);
    } else {
      int u = j0 - 64;
      return ldv8(&s.l1[c * 512 + m * 32 + u]) * (_Float16)sh0 * ldv8(&s.wmlp[m * 192 + 128 + u]);
    }
  };
  v16h am0[3], am1[3][3];
#pragma unroll
  for (int kt = 0; kt < 3; kt++) {
    int j0 = kt * 32 + kh * 8;
    am0[kt] = pack16(mid0seg(j0), mid0seg(j0 + 16));
#pragma unroll
    for (int c = 0; c < 3; c++) am1[c][kt] = pack16(mid1seg(c, j0), mid1seg(c, j0 + 16));
  }

  // -------- lin(mid) + sc(x) fused accumulation --------
  v8f o0acc[2] = { zero8(), zero8() };
  v8f o1acc[3] = { zero8(), zero8(), zero8() };
#pragma unroll
  for (int a = 0; a < 8; a++) {
    _Float16 sa = (_Float16)attrRow[a];
    { v16h t0 = ax00 * sa, t1 = ax01 * sa;
#pragma unroll
      for (int nt = 0; nt < 2; nt++) {
        o0acc[nt] = wmma16(t0, frag_b(W + BOFF_SCW0 + ((size_t)0 * 8 + a) * 32 + nt * 16, 256), o0acc[nt]);
        o0acc[nt] = wmma16(t1, frag_b(W + BOFF_SCW0 + ((size_t)(32) * 8 + a) * 32 + nt * 16, 256), o0acc[nt]);
      }
#pragma unroll
      for (int c = 0; c < 3; c++) {
        v16h tc = ax1[c] * sa;
        o1acc[c] = wmma16(tc, frag_b(W + BOFF_SCW1 + ((size_t)0 * 8 + a) * 16, 128), o1acc[c]);
      } }
#pragma unroll
    for (int kt = 0; kt < 3; kt++) {
      v16h tm = am0[kt] * sa;
#pragma unroll
      for (int nt = 0; nt < 2; nt++)
        o0acc[nt] = wmma16(tm, frag_b(W + BOFF_LINW0 + ((size_t)(kt * 32) * 8 + a) * 32 + nt * 16, 256), o0acc[nt]);
    }
#pragma unroll
    for (int c = 0; c < 3; c++)
#pragma unroll
      for (int kt = 0; kt < 3; kt++) {
        v16h tm = am1[c][kt] * sa;
        o1acc[c] = wmma16(tm, frag_b(W + BOFF_LINW1 + ((size_t)(kt * 32) * 8 + a) * 16, 128), o1acc[c]);
      }
  }
#pragma unroll
  for (int nt = 0; nt < 2; nt++) store_c_f(s.o0f, 32, nt * 16, o0acc[nt]);
#pragma unroll
  for (int c = 0; c < 3; c++) store_c_f(s.o1f + c * 256, 16, 0, o1acc[c]);
  __syncthreads();

  if (!IS_V) {
    // lane -> (edge me, head h); 40-dim dot with gathered q
    int me = lane >> 1, h = lane & 1;
    const float* q = qh + (size_t)s.dstI[me] * 80 + h * 40;
    float dot = 0.f;
#pragma unroll
    for (int p = 0; p < 40; p++) {
      float kv;
      if (p < 16) kv = s.o0f[me * 32 + h * 16 + p];
      else { int t = p - 16; int f1 = h * 24 + t; int w = f1 / 3, c = f1 % 3; kv = s.o1f[c * 256 + me * 16 + w]; }
      dot += q[p] * kv;
    }
    dot *= 0.15811388f; // 1/sqrt(40)
    if (active) {
      logits[(size_t)(e0 + me) * 2 + h] = dot;
      atomicMax(&mEnc[(size_t)s.dstI[me] * 2 + h], fenc(dot));
    }
  } else {
    for (int i = lane; i < 16 * 80; i += 32) {
      int me = i / 80, f = i % 80, h = f / 40, p = f % 40;
      float kv;
      if (p < 16) kv = s.o0f[me * 32 + h * 16 + p];
      else { int t = p - 16; int f1 = h * 24 + t; int w = f1 / 3, c = f1 % 3; kv = s.o1f[c * 256 + me * 16 + w]; }
      if (active) vh[(size_t)(e0 + me) * 80 + f] = (_Float16)kv;
    }
  }
}

// ---------------- softmax accumulate (pass B) ----------------
__global__ void softmax_acc_kernel(const int* __restrict__ edst, const float* __restrict__ logits,
                                   const unsigned* __restrict__ mEnc, const _Float16* __restrict__ vh,
                                   float* __restrict__ num, float* __restrict__ den) {
  size_t i = (size_t)blockIdx.x * blockDim.x + threadIdx.x;
  if (i >= (size_t)N_EDGES * 80) return;
  int e = (int)(i / 80), f = (int)(i % 80), h = f / 40;
  int n = edst[e];
  float mm = fdec(mEnc[(size_t)n * 2 + h]);
  float z = __expf(logits[(size_t)e * 2 + h] - mm);
  atomicAdd(&num[(size_t)n * 80 + f], z * (float)vh[i]);
  if ((f % 40) == 0) atomicAdd(&den[(size_t)n * 2 + h], z);
}

// ---------------- node finalize: normalize + LayerNorm + vector rescale ----------------
__global__ void finalize_kernel(const float* __restrict__ num, const float* __restrict__ den,
                                const float* __restrict__ ln_w, const float* __restrict__ ln_b,
                                float* __restrict__ out) {
  int n = blockIdx.x * blockDim.x + threadIdx.x;
  if (n >= N_NODES) return;
  float d0 = den[n * 2] + 1e-9f, d1 = den[n * 2 + 1] + 1e-9f;
  float oh[80];
#pragma unroll
  for (int f = 0; f < 80; f++) oh[f] = num[(size_t)n * 80 + f] / (f < 40 ? d0 : d1);
  float o0[32];
#pragma unroll
  for (int j = 0; j < 32; j++) { int h = j / 16; o0[j] = oh[h * 40 + (j & 15)]; }
  float mu = 0.f;
#pragma unroll
  for (int j = 0; j < 32; j++) mu += o0[j];
  mu *= (1.f / 32.f);
  float var = 0.f;
#pragma unroll
  for (int j = 0; j < 32; j++) { float d = o0[j] - mu; var += d * d; }
  var *= (1.f / 32.f);
  float inv = rsqrtf(var + 1e-5f);
#pragma unroll
  for (int j = 0; j < 32; j++) out[(size_t)n * 80 + j] = (o0[j] - mu) * inv * ln_w[j] + ln_b[j];
  float o1[48]; float fn = 0.f;
#pragma unroll
  for (int f1 = 0; f1 < 48; f1++) { int h = f1 / 24, t = f1 % 24; float v = oh[h * 40 + 16 + t]; o1[f1] = v; fn += v * v; }
  fn *= (1.f / 48.f);
  float r = rsqrtf(fn + 1e-5f);
#pragma unroll
  for (int w = 0; w < 16; w++) {
    float rs = ln_w[32 + w] * r;
#pragma unroll
    for (int c = 0; c < 3; c++) out[(size_t)n * 80 + 32 + w * 3 + c] = o1[w * 3 + c] * rs;
  }
}

// ---------------- host launcher ----------------
extern "C" void kernel_launch(void* const* d_in, const int* in_sizes, int n_in,
                              void* d_out, int out_size, void* d_ws, size_t ws_size,
                              hipStream_t stream) {
  (void)in_sizes; (void)n_in; (void)out_size; (void)ws_size;
  const float* feats = (const float*)d_in[0];
  const float* attrs = (const float*)d_in[1];
  const int*   esrc  = (const int*)d_in[2];
  const int*   edst  = (const int*)d_in[3];
  const float* esh   = (const float*)d_in[4];
  const float* erad  = (const float*)d_in[5];
  const float* etopo = (const float*)d_in[6];
  const float* ln_w  = (const float*)d_in[27];
  const float* ln_b  = (const float*)d_in[28];

  char* ws = (char*)d_ws;
  _Float16* Wf     = (_Float16*)(ws + OFF_W);
  float*    qh     = (float*)(ws + OFF_QH);
  float*    logits = (float*)(ws + OFF_LOG);
  _Float16* vh     = (_Float16*)(ws + OFF_VH);
  unsigned* mEnc   = (unsigned*)(ws + OFF_MENC);
  float*    den    = (float*)(ws + OFF_DEN);
  float*    num    = (float*)(ws + OFF_NUM);

  hipMemsetAsync(mEnc, 0, (size_t)N_NODES * 2 * 4, stream);
  hipMemsetAsync(den, 0, (size_t)N_NODES * 2 * 4, stream);
  hipMemsetAsync(num, 0, (size_t)N_NODES * 80 * 4, stream);

  const float S512 = 0.04419417382415922f;  // 1/sqrt(64*8)
  const float S256 = 0.0625f;               // 1/sqrt(32*8)
  const float S768 = 0.03608439182435161f;  // 1/sqrt(96*8)
  struct WCv { int in; int rows, cols, srcRows; float scale; size_t off; };
  const WCv tbl[] = {
    {7, 16384, 1, 16384, S512, WOFF_Q0},
    {8,  4096, 1,  4096, S256, WOFF_Q1},
    {9,  32768, 1, 32768, S512, WOFF_BR + BOFF_LIN0W0},
    {10,  8192, 1,  8192, S256, WOFF_BR + BOFF_LIN0W1},
    {11, 24576, 1, 24576, S768, WOFF_BR + BOFF_LINW0},
    {12, 12288, 1, 12288, S768, WOFF_BR + BOFF_LINW1},
    {13, 16384, 1, 16384, S512, WOFF_BR + BOFF_SCW0},
    {14,  4096, 1,  4096, S256, WOFF_BR + BOFF_SCW1},
    {15,    64, 64,   40, 1.f,  WOFF_BR + BOFF_FCW1},
    {16,  4096, 1,  4096, 1.f,  WOFF_BR + BOFF_FCW2},
    {17, 12288, 1, 12288, 1.f,  WOFF_BR + BOFF_FCW3},
    {18, 32768, 1, 32768, S512, WOFF_BR + BR_HALVES + BOFF_LIN0W0},
    {19,  8192, 1,  8192, S256, WOFF_BR + BR_HALVES + BOFF_LIN0W1},
    {20, 24576, 1, 24576, S768, WOFF_BR + BR_HALVES + BOFF_LINW0},
    {21, 12288, 1, 12288, S768, WOFF_BR + BR_HALVES + BOFF_LINW1},
    {22, 16384, 1, 16384, S512, WOFF_BR + BR_HALVES + BOFF_SCW0},
    {23,  4096, 1,  4096, S256, WOFF_BR + BR_HALVES + BOFF_SCW1},
    {24,    64, 64,   40, 1.f,  WOFF_BR + BR_HALVES + BOFF_FCW1},
    {25,  4096, 1,  4096, 1.f,  WOFF_BR + BR_HALVES + BOFF_FCW2},
    {26, 12288, 1, 12288, 1.f,  WOFF_BR + BR_HALVES + BOFF_FCW3},
  };
  for (const WCv& t : tbl) {
    int cnt = t.rows * t.cols;
    convert_kernel<<<(cnt + 255) / 256, 256, 0, stream>>>(
        (const float*)d_in[t.in], Wf + t.off, t.rows, t.cols, t.srcRows, t.scale);
  }

  q_kernel<<<(N_NODES / 16 + 3) / 4, 128, 0, stream>>>(feats, attrs, Wf, qh);

  const int edgeBlocks = (N_EDGES / 16 + 3) / 4;
  edge_kv_kernel<false><<<edgeBlocks, 128, 0, stream>>>(
      feats, attrs, esrc, edst, esh, erad, etopo, Wf + WOFF_BR, qh, mEnc, logits, vh);
  edge_kv_kernel<true><<<edgeBlocks, 128, 0, stream>>>(
      feats, attrs, esrc, edst, esh, erad, etopo, Wf + WOFF_BR + BR_HALVES, qh, mEnc, logits, vh);

  size_t tot = (size_t)N_EDGES * 80;
  softmax_acc_kernel<<<(unsigned)((tot + 255) / 256), 256, 0, stream>>>(edst, logits, mEnc, vh, num, den);

  finalize_kernel<<<(N_NODES + 255) / 256, 256, 0, stream>>>(num, den, ln_w, ln_b, (float*)d_out);
}
